// GanModel_18519898980388
// MI455X (gfx1250) — matrix operand order repaired and checked by reading, not compiled, not run on hardware
//
#include <hip/hip_runtime.h>
#include <cstdint>
#include <cstddef>

// ---------------------------------------------------------------------------
// MI455X (gfx1250) adaptive-GAN forward pass.
// Heavy convs = implicit GEMM on v_wmma_f32_16x16x32_f16 (f16 in, f32 acc),
// wave32, each wave owns a 16(oc) x 64(px) strip. Activations live in
// halo-padded NHWC f16 buffers (halo pre-zeroed), so the K-loop has NO
// boundary predication: one base address per tap, all fragment loads are
// immediate-offset global_load_b128, followed by back-to-back WMMAs.
// Weights are prepacked on-device into the CDNA5 A-fragment lane layout.
// ---------------------------------------------------------------------------

typedef __attribute__((ext_vector_type(16))) _Float16 v16h;
typedef __attribute__((ext_vector_type(8)))  _Float16 v8h;
typedef __attribute__((ext_vector_type(8)))  float    v8f;
typedef __attribute__((ext_vector_type(4)))  float    v4f;

static constexpr int Bn = 4, Hn = 257, Wn = 251;
// Padded activation geometry: y halo 2 (max padY), x halo 16 left + 21 right
// (max padX 14 for dilation-7 5x5, plus 5-column strip overrun) -> PW=288.
static constexpr int PHn = Hn + 4;    // 261
static constexpr int PWn = 288;
static constexpr int YOFF = 2, XOFF = 16;

__device__ __forceinline__ float siluf(float x) { return x / (1.0f + __expf(-x)); }

union AFrag16 { v16h v; v8h h[2]; };

// ---------------------------------------------------------------------------
// Zero-fill (vectorized) for padded buffers.
// ---------------------------------------------------------------------------
__global__ void k_zero(v4f* __restrict__ p, long n)
{
  long i = (long)blockIdx.x * blockDim.x + threadIdx.x;
  if (i < n) {
    v4f z = {0.f, 0.f, 0.f, 0.f};
    p[i] = z;
  }
}

// ---------------------------------------------------------------------------
// up1: conv 2->64 (channel 1 all-zero), 3x3 pad 1, SiLU, padded NHWC f16.
// ---------------------------------------------------------------------------
__global__ void k_up1(const float* __restrict__ spec,
                      const float* __restrict__ w,
                      const float* __restrict__ bias,
                      _Float16* __restrict__ outh)
{
  int tid = blockIdx.x * blockDim.x + threadIdx.x;
  const int total = Bn * Hn * Wn * 64;
  if (tid >= total) return;
  int oc = tid & 63;
  int p  = tid >> 6;
  int x  = p % Wn;
  int y  = (p / Wn) % Hn;
  int b  = p / (Wn * Hn);
  float acc = bias[oc];
#pragma unroll
  for (int ky = 0; ky < 3; ++ky) {
    int iy = y + ky - 1;
    if (iy < 0 || iy >= Hn) continue;
#pragma unroll
    for (int kx = 0; kx < 3; ++kx) {
      int ix = x + kx - 1;
      if (ix < 0 || ix >= Wn) continue;
      acc += spec[((size_t)b * Hn + iy) * Wn + ix] * w[(oc * 2 + 0) * 9 + ky * 3 + kx];
    }
  }
  outh[(((size_t)b * PHn + y + YOFF) * PWn + x + XOFF) * 64 + oc] = (_Float16)siluf(acc);
}

// ---------------------------------------------------------------------------
// Pack shared conv weights (OIHW f32) into WMMA A-fragment f16 layout.
// K order: K = tap*IC + ic (tap = ky*KW + kx); 32|IC so a 32-tile never
// straddles a tap. Lane L<16 holds K {0..7,16..23}, L>=16 holds {8..15,24..31}.
// ---------------------------------------------------------------------------
__global__ void k_pack_w(const float* __restrict__ w, _Float16* __restrict__ dst,
                         int OC, int IC, int KH, int KW, int OCT, int total)
{
  int idx = blockIdx.x * blockDim.x + threadIdx.x;
  if (idx >= total) return;
  int j = idx & 15;
  int L = (idx >> 4) & 31;
  int rest = idx >> 9;
  int octile = rest % OCT;
  int kt = rest / OCT;
  int m = L & 15;
  int kk = (L < 16) ? ((j < 8) ? j : j + 8) : ((j < 8) ? j + 8 : j + 16);
  int K = kt * 32 + kk;
  int t = K / IC, ic = K % IC;
  int ky = t / KW, kx = t % KW;
  int oc = octile * 16 + m;
  float v = (oc < OC) ? w[(((size_t)oc * IC + ic) * KH + ky) * KW + kx] : 0.0f;
  dst[idx] = (_Float16)v;
}

// Per-batch modulated adaptive kernels: src layout (IC,OC,3,3) scaled by
// wa[b,ic]; rows oc>=OC zero-filled (final layer OC==1).
__global__ void k_pack_adapt(const float* __restrict__ w, const float* __restrict__ wa,
                             _Float16* __restrict__ dst, int OC, int IC, int OCT,
                             int per, int total)
{
  int idx = blockIdx.x * blockDim.x + threadIdx.x;
  if (idx >= total) return;
  int b = idx / per;
  int e = idx % per;
  int j = e & 15;
  int L = (e >> 4) & 31;
  int rest = e >> 9;
  int octile = rest % OCT;
  int kt = rest / OCT;
  int m = L & 15;
  int kk = (L < 16) ? ((j < 8) ? j : j + 8) : ((j < 8) ? j + 8 : j + 16);
  int K = kt * 32 + kk;
  int t = K / IC, ic = K % IC;
  int ky = t / 3, kx = t % 3;
  int oc = octile * 16 + m;
  float v = 0.0f;
  if (oc < OC) v = w[(((size_t)ic * OC + oc) * 3 + ky) * 3 + kx] * wa[b * IC + ic];
  dst[idx] = (_Float16)v;
}

// ---------------------------------------------------------------------------
// Generic WMMA implicit-GEMM conv over padded NHWC f16 input.
// MODE 0: +bias, SiLU, padded f16 NHWC out               (up2)
// MODE 1: +bias, SiLU, *res[b,d]*0.25, dense f32 NHWC    (dilated, accum)
// MODE 2: +bias, SiLU, dense f32 [B,W,COND] out          (gi, KH=257,KW=1)
// MODE 3: +bmod[b,oc], padded f16 NHWC out               (adaptive 0..2)
// MODE 4: +bmod[b,0],  dense f32 [B,1,H,W] out           (adaptive 3, OC=1)
// ---------------------------------------------------------------------------
template <int MODE, int ICc>
__global__ __launch_bounds__(32)
void k_wmma_conv(const _Float16* __restrict__ packA,
                 const _Float16* __restrict__ inh,
                 const float* __restrict__ bias,
                 const float* __restrict__ resm,
                 float* __restrict__ outf,
                 _Float16* __restrict__ outh,
                 int OH, int OW, int OC,
                 int KH, int KW, int padY, int padX, int dilX,
                 int OCT, int perBatchA, int dIdx, int accum)
{
  constexpr int ICT = ICc / 32;
  __builtin_amdgcn_s_wait_tensorcnt(0);  // CDNA5 split dep-counter (no-op here)

  const int lane = threadIdx.x;
  const int octile = blockIdx.y;
  const int ptile = blockIdx.x;
  const int owt = (OW + 63) >> 6;
  const int b  = ptile / (OH * owt);
  const int r0 = ptile % (OH * owt);
  const int oy = r0 / owt;
  const int xbase = (r0 % owt) << 6;
  const int n  = lane & 15;   // column within a 16-px tile
  const int kg = lane >> 4;   // K half-group
  const int x0 = xbase + n;   // pixel of tile 0 for this lane

  const size_t aStride = (size_t)OCT * 512;  // halves between consecutive K-tiles
  const _Float16* aPtr = packA + (size_t)b * perBatchA
                       + (size_t)octile * 512 + (size_t)lane * 16;

  v8f acc[4];
#pragma unroll
  for (int j = 0; j < 4; ++j)
#pragma unroll
    for (int r = 0; r < 8; ++r) acc[j][r] = 0.f;

  for (int ky = 0; ky < KH; ++ky) {
    const int iyp = oy + ky - padY + YOFF;               // always in [0,PHn)
    const _Float16* rowBase =
        inh + ((size_t)(b * PHn + iyp) * PWn) * ICc + kg * 8;
    for (int kx = 0; kx < KW; ++kx) {
      const int ixp = x0 + kx * dilX - padX + XOFF;      // always in [0,PWn)
      const _Float16* bBase = rowBase + (size_t)ixp * ICc;
      // Warm caches for the next A panel (global_prefetch_b8).
      __builtin_prefetch((const void*)(aPtr + ICT * aStride), 0, 1);
#pragma unroll
      for (int ict = 0; ict < ICT; ++ict) {
        AFrag16 a;
        const v8h* ap = (const v8h*)aPtr;
        a.h[0] = ap[0];
        a.h[1] = ap[1];
        aPtr += aStride;
#pragma unroll
        for (int j = 0; j < 4; ++j) {
          AFrag16 bf;  // constexpr offsets -> immediate-offset b128 loads
          const v8h* q = (const v8h*)(bBase + ict * 32 + j * 16 * ICc);
          bf.h[0] = q[0];
          bf.h[1] = q[2];
          acc[j] = __builtin_amdgcn_wmma_f32_16x16x32_f16(false, a.v, false, bf.v,
                                                          (short)0, acc[j], false, false);
        }
      }
    }
  }

  // C layout: element (r, lane) = C[m = r + 8*kg, n]; rows r are consecutive
  // output channels for a fixed lane -> NHWC stores are one b128 per tile.
  const int ocb = (octile << 4) + (kg << 3);
#pragma unroll
  for (int j = 0; j < 4; ++j) {
    const int xcol = x0 + 16 * j;
    if (xcol >= OW) continue;
    if (MODE == 0 || MODE == 3) {
      _Float16* o = outh + (((size_t)(b * PHn + oy + YOFF)) * PWn + xcol + XOFF) * OC + ocb;
      v8h st;
#pragma unroll
      for (int r = 0; r < 8; ++r) {
        float v = acc[j][r];
        if (MODE == 0) v = siluf(v + bias[ocb + r]);
        else           v = v + bias[b * OC + ocb + r];
        st[r] = (_Float16)v;
      }
      *(v8h*)o = st;
    } else if (MODE == 1) {
      const float rs = resm[((size_t)(b * 4 + dIdx) * OH + oy) * OW + xcol] * 0.25f;
      float* o = outf + ((size_t)(b * OH + oy) * OW + xcol) * OC + ocb;
#pragma unroll
      for (int r = 0; r < 8; ++r) {
        float v = siluf(acc[j][r] + bias[ocb + r]) * rs;
        o[r] = accum ? (o[r] + v) : v;
      }
    } else if (MODE == 2) {
      float* o = outf + ((size_t)(b * OW + xcol)) * OC + ocb;
#pragma unroll
      for (int r = 0; r < 8; ++r) o[r] = siluf(acc[j][r] + bias[ocb + r]);
    } else {  // MODE 4: only oc==0 exists
      if (ocb == 0)
        outf[((size_t)b * OH + oy) * OW + xcol] = acc[j][0] + bias[b * OC + 0];
    }
  }
}

// ---------------------------------------------------------------------------
// res conv3d over padded f1h (halo zeros == conv zero-padding, no branches).
// ---------------------------------------------------------------------------
__global__ void k_res3d(const _Float16* __restrict__ f1h,
                        const float* __restrict__ rw,
                        const float* __restrict__ rb,
                        float* __restrict__ res)
{
  int tid = blockIdx.x * blockDim.x + threadIdx.x;
  const int total = Bn * 4 * Hn * Wn;
  if (tid >= total) return;
  int x = tid % Wn;
  int y = (tid / Wn) % Hn;
  int d = (tid / (Wn * Hn)) & 3;
  int b = tid / (Wn * Hn * 4);
  float acc = rb[d];
#pragma unroll
  for (int ky = 0; ky < 5; ++ky) {
#pragma unroll
    for (int kx = 0; kx < 5; ++kx) {
      const _Float16* ip =
          f1h + (((size_t)b * PHn + y + ky - 2 + YOFF) * PWn + x + kx - 2 + XOFF) * 128;
      const float* wp = rw + (size_t)d * 128 * 25 + ky * 5 + kx;
#pragma unroll 4
      for (int ic = 0; ic < 128; ++ic)
        acc += (float)ip[ic] * wp[(size_t)ic * 25];
    }
  }
  res[tid] = acc;
}

// ---------------------------------------------------------------------------
// Channel-wise L2 normalize of o (dense f32 NHWC) -> padded f16 NHWC.
// ---------------------------------------------------------------------------
__global__ void k_normalize(const float* __restrict__ o, _Float16* __restrict__ f2h)
{
  __shared__ float red[128];
  int p = blockIdx.x;  // pixel (b,y,x)
  int c = threadIdx.x;
  int x = p % Wn;
  int y = (p / Wn) % Hn;
  int b = p / (Wn * Hn);
  float v = o[(size_t)p * 128 + c];
  red[c] = v * v;
  __syncthreads();
  for (int s = 64; s > 0; s >>= 1) {
    if (c < s) red[c] += red[c + s];
    __syncthreads();
  }
  float nrm = fmaxf(sqrtf(red[0]), 1e-12f);
  f2h[(((size_t)b * PHn + y + YOFF) * PWn + x + XOFF) * 128 + c] = (_Float16)(v / nrm);
}

// ---------------------------------------------------------------------------
// Tiny GEMVs + softmax attention.
// ---------------------------------------------------------------------------
__global__ void k_q(const float* __restrict__ cond_in, const float* __restrict__ wq,
                    const float* __restrict__ bq, float* __restrict__ q)
{
  int tid = blockIdx.x * blockDim.x + threadIdx.x;
  if (tid >= Bn * 256) return;
  int c = tid & 255, b = tid >> 8;
  float a = bq[c];
  for (int k = 0; k < 256; ++k) a += cond_in[b * 256 + k] * wq[c * 256 + k];
  q[tid] = a;
}

__global__ void k_kv(const float* __restrict__ gi,
                     const float* __restrict__ wk, const float* __restrict__ bk,
                     const float* __restrict__ wv, const float* __restrict__ bv,
                     float* __restrict__ kb, float* __restrict__ vb)
{
  int tid = blockIdx.x * blockDim.x + threadIdx.x;
  if (tid >= Bn * Wn * 256) return;
  int c = tid & 255;
  int bw = tid >> 8;
  const float* g = gi + (size_t)bw * 256;
  float ak = bk[c], av = bv[c];
  for (int j = 0; j < 256; ++j) {
    float gv = g[j];
    ak += gv * wk[c * 256 + j];
    av += gv * wv[c * 256 + j];
  }
  kb[tid] = ak;
  vb[tid] = av;
}

__global__ void k_attn(const float* __restrict__ q, const float* __restrict__ kb,
                       const float* __restrict__ vb, float* __restrict__ outa)
{
  __shared__ float sc[256];
  __shared__ float red[256];
  int bh = blockIdx.x;
  int b = bh >> 1, h = bh & 1;
  int tid = threadIdx.x;
  const float* qh = q + b * 256 + h * 128;
  float s = -1e30f;
  if (tid < Wn) {
    const float* kr = kb + ((size_t)(b * Wn + tid)) * 256 + h * 128;
    float acc = 0.f;
    for (int d = 0; d < 128; ++d) acc += qh[d] * kr[d];
    s = acc * 0.08838834764831845f;  // 1/sqrt(128)
  }
  sc[tid] = s;
  red[tid] = s;
  __syncthreads();
  for (int st = 128; st > 0; st >>= 1) {
    if (tid < st) red[tid] = fmaxf(red[tid], red[tid + st]);
    __syncthreads();
  }
  float mx = red[0];
  __syncthreads();
  float e = (tid < Wn) ? __expf(sc[tid] - mx) : 0.f;
  sc[tid] = e;
  red[tid] = e;
  __syncthreads();
  for (int st = 128; st > 0; st >>= 1) {
    if (tid < st) red[tid] += red[tid + st];
    __syncthreads();
  }
  float inv = 1.0f / red[0];
  if (tid < 128) {
    float acc = 0.f;
    for (int w2 = 0; w2 < Wn; ++w2)
      acc += sc[w2] * vb[((size_t)(b * Wn + w2)) * 256 + h * 128 + tid];
    outa[b * 256 + h * 128 + tid] = acc * inv;
  }
}

__global__ void k_outproj(const float* __restrict__ ain, const float* __restrict__ wo,
                          const float* __restrict__ bo, float* __restrict__ cond)
{
  int tid = blockIdx.x * blockDim.x + threadIdx.x;
  if (tid >= Bn * 256) return;
  int c = tid & 255, b = tid >> 8;
  float a = bo[c];
  for (int j = 0; j < 256; ++j) a += ain[b * 256 + j] * wo[c * 256 + j];
  cond[tid] = a;
}

// wa[b,ic] and bmod[b,oc]=b[oc]*ba[b,oc] from cat(cond, pos_emb[l]).
__global__ void k_waba(const float* __restrict__ cond, const float* __restrict__ pos,
                       const float* __restrict__ wa_w, const float* __restrict__ wa_b,
                       const float* __restrict__ ba_w, const float* __restrict__ ba_b,
                       const float* __restrict__ bconv, int IC, int OC,
                       float* __restrict__ wa, float* __restrict__ bmod)
{
  int tid = blockIdx.x * blockDim.x + threadIdx.x;
  int per = IC + OC;
  if (tid >= Bn * per) return;
  int b = tid / per;
  int j = tid % per;
  const float* row;
  float acc;
  if (j < IC) { row = wa_w + (size_t)j * 512; acc = wa_b[j]; }
  else        { row = ba_w + (size_t)(j - IC) * 512; acc = ba_b[j - IC]; }
  for (int k = 0; k < 256; ++k) acc += cond[b * 256 + k] * row[k];
  for (int k = 0; k < 256; ++k) acc += pos[k] * row[256 + k];
  if (j < IC) wa[b * IC + j] = acc;
  else        bmod[b * OC + (j - IC)] = bconv[j - IC] * acc;
}

// ---------------------------------------------------------------------------
// Host orchestration.
// ---------------------------------------------------------------------------
extern "C" void kernel_launch(void* const* d_in, const int* in_sizes, int n_in,
                              void* d_out, int out_size, void* d_ws, size_t ws_size,
                              hipStream_t stream)
{
  (void)in_sizes; (void)n_in; (void)out_size; (void)ws_size;

  const float* spec   = (const float*)d_in[0];
  const float* condIn = (const float*)d_in[1];
  const float* up1_w  = (const float*)d_in[2];
  const float* up1_b  = (const float*)d_in[3];
  const float* up2_w  = (const float*)d_in[4];
  const float* up2_b  = (const float*)d_in[5];
  const float* dil_w[4] = {(const float*)d_in[6], (const float*)d_in[7],
                           (const float*)d_in[8], (const float*)d_in[9]};
  const float* dil_b[4] = {(const float*)d_in[10], (const float*)d_in[11],
                           (const float*)d_in[12], (const float*)d_in[13]};
  const float* res_w = (const float*)d_in[14];
  const float* res_b = (const float*)d_in[15];
  const float* gi_w  = (const float*)d_in[16];
  const float* gi_b  = (const float*)d_in[17];
  const float* wq = (const float*)d_in[18];
  const float* bq = (const float*)d_in[19];
  const float* wk = (const float*)d_in[20];
  const float* bk = (const float*)d_in[21];
  const float* wv = (const float*)d_in[22];
  const float* bv = (const float*)d_in[23];
  const float* wo = (const float*)d_in[24];
  const float* bo = (const float*)d_in[25];
  const float* pos_emb = (const float*)d_in[26];

  char* ws = (char*)d_ws;
  size_t cur = 0;
  auto alloc = [&](size_t bytes) -> void* {
    void* p = ws + cur;
    cur = (cur + bytes + 255) & ~(size_t)255;
    return p;
  };
  const size_t NPIX = (size_t)Bn * Hn * Wn;            // dense pixels
  const size_t PPIX = (size_t)Bn * PHn * PWn;          // padded pixels
  _Float16* i0h  = (_Float16*)alloc(PPIX * 64 * 2);    // padded NHWC
  _Float16* f1h  = (_Float16*)alloc(PPIX * 128 * 2);   // padded NHWC
  float*    obuf = (float*)alloc(NPIX * 128 * 4);      // dense NHWC accumulator
  float*    res  = (float*)alloc(NPIX * 4 * 4);
  _Float16* ping = (_Float16*)alloc(PPIX * 128 * 2);   // padded NHWC
  _Float16* pong = (_Float16*)alloc(PPIX * 128 * 2);   // padded NHWC
  float*    gi   = (float*)alloc((size_t)Bn * Wn * 256 * 4);
  float*    qbuf = (float*)alloc(Bn * 256 * 4);
  float*    attn = (float*)alloc(Bn * 256 * 4);
  float*    cond = (float*)alloc(Bn * 256 * 4);
  float*    kbuf = (float*)alloc((size_t)Bn * Wn * 256 * 4);
  float*    vbuf = (float*)alloc((size_t)Bn * Wn * 256 * 4);
  float*    wa   = (float*)alloc(Bn * 128 * 4);
  float*    bmod = (float*)alloc(Bn * 128 * 4);
  _Float16* pUp2 = (_Float16*)alloc((size_t)18 * 8 * 512 * 2);
  _Float16* pDil = (_Float16*)alloc((size_t)4 * 100 * 8 * 512 * 2);
  _Float16* pGi  = (_Float16*)alloc((size_t)1028 * 16 * 512 * 2);
  _Float16* pAd  = (_Float16*)alloc((size_t)Bn * 36 * 8 * 512 * 2);

  auto zero = [&](void* p, size_t bytes) {
    long n = (long)(bytes / 16);
    k_zero<<<(unsigned)((n + 255) / 256), 256, 0, stream>>>((v4f*)p, n);
  };
  // Pre-zero padded buffers (halo must read as conv zero-padding).
  zero(i0h,  PPIX * 64 * 2);
  zero(f1h,  PPIX * 128 * 2);
  zero(ping, PPIX * 128 * 2);
  zero(pong, PPIX * 128 * 2);

  auto cgrid = [&](int OH, int OW, int OC) {
    return dim3((unsigned)(Bn * OH * ((OW + 63) / 64)), (unsigned)((OC + 15) / 16));
  };
  dim3 blk(32);

  // 1. up1 (direct) -> i0h padded NHWC f16
  {
    int total = (int)(NPIX * 64);
    k_up1<<<(total + 255) / 256, 256, 0, stream>>>(spec, up1_w, up1_b, i0h);
  }

  // 2. up2 via WMMA: 64 -> 128, 3x3 pad1, SiLU -> f1h
  {
    int total = 18 * 8 * 512;
    k_pack_w<<<(total + 255) / 256, 256, 0, stream>>>(up2_w, pUp2, 128, 64, 3, 3, 8, total);
    k_wmma_conv<0, 64><<<cgrid(Hn, Wn, 128), blk, 0, stream>>>(
        pUp2, i0h, up2_b, nullptr, nullptr, f1h,
        Hn, Wn, 128, 3, 3, 1, 1, 1, 8, 0, 0, 0);
  }

  // 3. res conv3d (direct, reads padded f1h)
  {
    int total = Bn * 4 * Hn * Wn;
    k_res3d<<<(total + 255) / 256, 256, 0, stream>>>(f1h, res_w, res_b, res);
  }

  // 4. dilated 5x5 convs via WMMA: o = mean_d(silu(dil_d) * res_d), dense NHWC
  for (int d = 0; d < 4; ++d) {
    int dd = 2 * d + 1;  // dilation 1,3,5,7
    int total = 100 * 8 * 512;
    _Float16* pd = pDil + (size_t)d * total;
    k_pack_w<<<(total + 255) / 256, 256, 0, stream>>>(dil_w[d], pd, 128, 128, 5, 5, 8, total);
    k_wmma_conv<1, 128><<<cgrid(Hn, Wn, 128), blk, 0, stream>>>(
        pd, f1h, dil_b[d], res, obuf, nullptr,
        Hn, Wn, 128, 5, 5, 2, 2 * dd, dd, 8, 0, d, d > 0 ? 1 : 0);
  }

  // 5. channel-normalize -> f2 (ping, padded NHWC f16)
  k_normalize<<<(unsigned)NPIX, 128, 0, stream>>>(obuf, ping);

  // 6. gi via WMMA: full-height (257x1) valid conv, SiLU -> gi [B,W,256]
  {
    int total = 1028 * 16 * 512;
    k_pack_w<<<(total + 255) / 256, 256, 0, stream>>>(gi_w, pGi, 256, 128, 257, 1, 16, total);
    k_wmma_conv<2, 128><<<cgrid(1, Wn, 256), blk, 0, stream>>>(
        pGi, f1h, gi_b, nullptr, gi, nullptr,
        1, Wn, 256, 257, 1, 0, 0, 1, 16, 0, 0, 0);
  }

  // 7. cross-attention (tiny)
  k_q<<<(Bn * 256 + 255) / 256, 256, 0, stream>>>(condIn, wq, bq, qbuf);
  {
    int total = Bn * Wn * 256;
    k_kv<<<(total + 255) / 256, 256, 0, stream>>>(gi, wk, bk, wv, bv, kbuf, vbuf);
  }
  k_attn<<<Bn * 2, 256, 0, stream>>>(qbuf, kbuf, vbuf, attn);
  k_outproj<<<(Bn * 256 + 255) / 256, 256, 0, stream>>>(attn, wo, bo, cond);

  // 8. adaptive conv stack (per-sample kernels via per-batch packed A)
  const int ICs[4] = {128, 128, 128, 64};
  const int OCs[4] = {128, 128, 64, 1};
  _Float16* ins[4]  = {ping, pong, ping, pong};
  _Float16* outs[3] = {pong, ping, pong};
  for (int l = 0; l < 4; ++l) {
    const float* aw   = (const float*)d_in[27 + 6 * l + 0];
    const float* ab   = (const float*)d_in[27 + 6 * l + 1];
    const float* wa_w = (const float*)d_in[27 + 6 * l + 2];
    const float* wa_b = (const float*)d_in[27 + 6 * l + 3];
    const float* ba_w = (const float*)d_in[27 + 6 * l + 4];
    const float* ba_b = (const float*)d_in[27 + 6 * l + 5];
    int IC = ICs[l], OC = OCs[l];
    int OCT = (OC + 15) / 16;
    int KT = (IC * 9) / 32;
    int per = KT * OCT * 512;
    int totw = Bn * (IC + OC);
    k_waba<<<(totw + 255) / 256, 256, 0, stream>>>(cond, pos_emb + l * 256, wa_w, wa_b,
                                                   ba_w, ba_b, ab, IC, OC, wa, bmod);
    int totp = Bn * per;
    k_pack_adapt<<<(totp + 255) / 256, 256, 0, stream>>>(aw, wa, pAd, OC, IC, OCT, per, totp);
    if (l < 3) {
      k_wmma_conv<3, 128><<<cgrid(Hn, Wn, OC), blk, 0, stream>>>(
          pAd, ins[l], bmod, nullptr, nullptr, outs[l],
          Hn, Wn, OC, 3, 3, 1, 1, 1, OCT, per, 0, 0);
    } else {
      k_wmma_conv<4, 64><<<cgrid(Hn, Wn, 1), blk, 0, stream>>>(
          pAd, ins[l], bmod, nullptr, (float*)d_out, nullptr,
          Hn, Wn, 1, 3, 3, 1, 1, 1, 1, per, 0, 0);
    }
  }
}
